// GeoTransformer_15513421873270
// MI455X (gfx1250) — matrix operand order repaired and checked by reading, not compiled, not run on hardware
//
#include <hip/hip_runtime.h>
#include <hip/hip_bf16.h>
#include <math.h>

// ---------------------------------------------------------------------------
// GeoTransformer-style pipeline for MI455X (gfx1250, wave32, WMMA).
//   N_REF=N_SRC=8192, D=256, rows M=16384.
//   Big GEMMs (25.8 GFLOP) run on v_wmma_f32_16x16x32_bf16 with fused
//   instance-norm/ReLU epilogues; pooling fused as deterministic partial sums.
// ---------------------------------------------------------------------------

#define N_REF   8192
#define N_SRC   8192
#define NPTS    16384
#define DFEAT   256
#define MROWS   16384
#define K1RAW   520
#define K1PAD   544     // 17 * 32
#define N1      256
#define N2      512
#define N3      1024
#define NCHUNK  8

typedef __attribute__((ext_vector_type(16))) __bf16 v16bf;
typedef __attribute__((ext_vector_type(8)))  float  v8f;

union Frag32 {            // 32 bytes: one WMMA bf16 A/B fragment per lane
    uint4 q[2];
    v16bf v;
};

__device__ __forceinline__ unsigned short f2bf(float x) {
    union { float f; unsigned int u; } c; c.f = x;
    unsigned int u = c.u;
    unsigned int r = u + 0x7FFFu + ((u >> 16) & 1u);   // round-to-nearest-even
    return (unsigned short)(r >> 16);
}

// ---------------------------------------------------------------------------
// 1) Row-wise L2 normalization of features (one block per row)
// ---------------------------------------------------------------------------
__global__ __launch_bounds__(256) void l2norm_kernel(const float* __restrict__ in,
                                                     float* __restrict__ out) {
    __shared__ float red[256];
    const int r = blockIdx.x, t = threadIdx.x;
    float v = in[(size_t)r * DFEAT + t];
    red[t] = v * v;
    __syncthreads();
    for (int s = 128; s > 0; s >>= 1) {
        if (t < s) red[t] += red[t + s];
        __syncthreads();
    }
    float nrm = fmaxf(sqrtf(red[0]), 1e-12f);
    out[(size_t)r * DFEAT + t] = v / nrm;
}

// ---------------------------------------------------------------------------
// 2) Point normalization: center by mean, scale by max norm (single block)
// ---------------------------------------------------------------------------
__global__ __launch_bounds__(1024) void ptsnorm_kernel(const float* __restrict__ refp,
                                                       const float* __restrict__ srcp,
                                                       float* __restrict__ ptsN) {
    __shared__ float rx[1024], ry[1024], rz[1024];
    const int t = threadIdx.x;
    float sx = 0.f, sy = 0.f, sz = 0.f;
    for (int i = t; i < NPTS; i += 1024) {
        const float* p = (i < N_REF) ? (refp + (size_t)i * 3)
                                     : (srcp + (size_t)(i - N_REF) * 3);
        sx += p[0]; sy += p[1]; sz += p[2];
    }
    rx[t] = sx; ry[t] = sy; rz[t] = sz;
    __syncthreads();
    for (int s = 512; s > 0; s >>= 1) {
        if (t < s) { rx[t] += rx[t + s]; ry[t] += ry[t + s]; rz[t] += rz[t + s]; }
        __syncthreads();
    }
    const float mx = rx[0] / (float)NPTS, my = ry[0] / (float)NPTS, mz = rz[0] / (float)NPTS;
    __syncthreads();                       // done reading before reuse of rx
    float mxn = 0.f;
    for (int i = t; i < NPTS; i += 1024) {
        const float* p = (i < N_REF) ? (refp + (size_t)i * 3)
                                     : (srcp + (size_t)(i - N_REF) * 3);
        float dx = p[0] - mx, dy = p[1] - my, dz = p[2] - mz;
        mxn = fmaxf(mxn, sqrtf(dx * dx + dy * dy + dz * dz));
    }
    rx[t] = mxn;
    __syncthreads();
    for (int s = 512; s > 0; s >>= 1) {
        if (t < s) rx[t] = fmaxf(rx[t], rx[t + s]);
        __syncthreads();
    }
    const float inv = 1.0f / rx[0];
    for (int i = t; i < NPTS; i += 1024) {
        const float* p = (i < N_REF) ? (refp + (size_t)i * 3)
                                     : (srcp + (size_t)(i - N_REF) * 3);
        ptsN[(size_t)i * 3 + 0] = (p[0] - mx) * inv;
        ptsN[(size_t)i * 3 + 1] = (p[1] - my) * inv;
        ptsN[(size_t)i * 3 + 2] = (p[2] - mz) * inv;
    }
}

// ---------------------------------------------------------------------------
// 3) Chunked min/argmin over pairwise distances (LDS-tiled, first-min kept)
// ---------------------------------------------------------------------------
__global__ __launch_bounds__(128) void minargmin_kernel(const float* __restrict__ Apts,
                                                        const float* __restrict__ Bpts,
                                                        int nA, int nB,
                                                        float* __restrict__ partD,
                                                        int* __restrict__ partI) {
    __shared__ float sb[512 * 3];
    const int i = blockIdx.x * 128 + threadIdx.x;
    const float ax = Apts[(size_t)i * 3], ay = Apts[(size_t)i * 3 + 1], az = Apts[(size_t)i * 3 + 2];
    const float aa = ax * ax + ay * ay + az * az;
    const int chunkLen = nB / NCHUNK;
    const int j0 = blockIdx.y * chunkLen;
    float best = INFINITY; int bestI = 0;
    for (int jt = 0; jt < chunkLen; jt += 512) {
        for (int u = threadIdx.x; u < 512 * 3; u += 128)
            sb[u] = Bpts[(size_t)(j0 + jt) * 3 + u];
        __syncthreads();
        #pragma unroll 4
        for (int j = 0; j < 512; ++j) {
            float bx = sb[j * 3], by = sb[j * 3 + 1], bz = sb[j * 3 + 2];
            float bb = bx * bx + by * by + bz * bz;
            float d2 = aa + bb - 2.0f * (ax * bx + ay * by + az * bz);
            float d = sqrtf(fmaxf(d2, 1e-12f));
            if (d < best) { best = d; bestI = j0 + jt + j; }   // first occurrence
        }
        __syncthreads();
    }
    partD[(size_t)blockIdx.y * nA + i] = best;
    partI[(size_t)blockIdx.y * nA + i] = bestI;
}

__global__ void minmerge_kernel(const float* __restrict__ partD,
                                const int* __restrict__ partI, int nA,
                                float* __restrict__ outD, int* __restrict__ outI) {
    const int i = blockIdx.x * 256 + threadIdx.x;
    if (i >= nA) return;
    float best = partD[i]; int bi = partI[i];
    for (int c = 1; c < NCHUNK; ++c) {           // ascending chunk order ->
        float d = partD[(size_t)c * nA + i];     // strict < keeps first min
        if (d < best) { best = d; bi = partI[(size_t)c * nA + i]; }
    }
    outD[i] = best; outI[i] = bi;
}

// ---------------------------------------------------------------------------
// 4) Build bf16 feat matrix (16384 x 544, K-padded) + correspondence scores
// ---------------------------------------------------------------------------
__global__ __launch_bounds__(256) void featbuild_kernel(const float* __restrict__ rfN,
                                                        const float* __restrict__ sfN,
                                                        const float* __restrict__ ptsN,
                                                        const float* __restrict__ dRef,
                                                        const int* __restrict__ iRef,
                                                        const float* __restrict__ dSrc,
                                                        const int* __restrict__ iSrc,
                                                        unsigned short* __restrict__ featB) {
    __shared__ float red[256];
    const int row = blockIdx.x, t = threadIdx.x;
    const float *own, *oth, *pown, *poth;
    float d;
    if (row < N_REF) {
        int j = iRef[row];
        own = rfN + (size_t)row * DFEAT;  oth = sfN + (size_t)j * DFEAT;
        pown = ptsN + (size_t)row * 3;    poth = ptsN + (size_t)(N_REF + j) * 3;
        d = dRef[row];
    } else {
        int j = row - N_REF, k = iSrc[j];
        own = sfN + (size_t)j * DFEAT;    oth = rfN + (size_t)k * DFEAT;
        pown = ptsN + (size_t)(N_REF + j) * 3; poth = ptsN + (size_t)k * 3;
        d = dSrc[j];
    }
    float a = own[t], b = oth[t];
    unsigned short* frow = featB + (size_t)row * K1PAD;
    frow[t]         = f2bf(a);
    frow[DFEAT + t] = f2bf(b);
    red[t] = a * b;                                    // score = <own, oth>
    __syncthreads();
    for (int s = 128; s > 0; s >>= 1) {
        if (t < s) red[t] += red[t + s];
        __syncthreads();
    }
    if (t == 0) {
        frow[512] = f2bf(pown[0]); frow[513] = f2bf(pown[1]); frow[514] = f2bf(pown[2]);
        frow[515] = f2bf(poth[0]); frow[516] = f2bf(poth[1]); frow[517] = f2bf(poth[2]);
        frow[518] = f2bf(d);
        frow[519] = f2bf(red[0]);
    }
    if (t >= 232) frow[288 + t] = 0;                   // zero-pad cols 520..543
}

// ---------------------------------------------------------------------------
// 5) Weight convert+transpose: fp32 (K x N) -> bf16 (N x Kpad), zero padded
// ---------------------------------------------------------------------------
__global__ void wconv_kernel(const float* __restrict__ W, unsigned short* __restrict__ Wt,
                             int K, int Kpad, int N) {
    size_t idx = (size_t)blockIdx.x * 256 + threadIdx.x;
    if (idx >= (size_t)N * Kpad) return;
    int n = (int)(idx / Kpad), k = (int)(idx % Kpad);
    Wt[idx] = (k < K) ? f2bf(W[(size_t)k * N + n]) : (unsigned short)0;
}

// ---------------------------------------------------------------------------
// 6) WMMA GEMM (M x K) x (K x N) with fused bias + instance-norm + ReLU
//    One block owns ALL N-tiles of its 16-row group (norm needs full rows).
//    8 waves; each wave computes N/128 16x16 tiles via v_wmma_f32_16x16x32_bf16.
// ---------------------------------------------------------------------------
template <int N, int K>
__global__ __launch_bounds__(256) void gemm_instnorm_relu(const unsigned short* __restrict__ A,
                                                          const unsigned short* __restrict__ Bt,
                                                          const float* __restrict__ bias,
                                                          unsigned short* __restrict__ out) {
    constexpr int NT_PER_WAVE = N / 16 / 8;
    __shared__ float st[16][N];
    __shared__ float red[2][16][16];
    __shared__ float mrow[16], rrow[16];

    const int tid = threadIdx.x;
    const int wave = tid >> 5, lane = tid & 31;
    const int h = lane >> 4, ln = lane & 15;
    const int mTile = blockIdx.x;

    const unsigned short* arow = A + (size_t)(mTile * 16 + ln) * K;

    v8f acc[NT_PER_WAVE];
    #pragma unroll
    for (int i = 0; i < NT_PER_WAVE; ++i) { v8f z = {}; acc[i] = z; }

    for (int k0 = 0; k0 < K; k0 += 32) {
        Frag32 af;                                    // A: 16x32 bf16 fragment
        af.q[0] = *(const uint4*)(arow + k0 + 16 * h);
        af.q[1] = *(const uint4*)(arow + k0 + 16 * h + 8);
        #pragma unroll
        for (int nt = 0; nt < NT_PER_WAVE; ++nt) {
            const int ntile = wave * NT_PER_WAVE + nt;
            const unsigned short* brow = Bt + (size_t)(ntile * 16 + ln) * K + k0 + 16 * h;
            Frag32 bf;                                // B: 32x16 bf16 fragment
            bf.q[0] = *(const uint4*)(brow);
            bf.q[1] = *(const uint4*)(brow + 8);
            acc[nt] = __builtin_amdgcn_wmma_f32_16x16x32_bf16(
                false, af.v, false, bf.v, (short)0, acc[nt], false, false);
        }
    }
    #pragma unroll
    for (int nt = 0; nt < NT_PER_WAVE; ++nt) {        // stage D + bias into LDS
        const int col = (wave * NT_PER_WAVE + nt) * 16 + ln;
        const float b = bias[col];
        #pragma unroll
        for (int r = 0; r < 8; ++r) st[r + 8 * h][col] = acc[nt][r] + b;
    }
    __syncthreads();
    {   // instance norm: 16 threads per row
        const int r = tid >> 4, sub = tid & 15;
        float s = 0.f, q = 0.f;
        for (int c = sub; c < N; c += 16) { float v = st[r][c]; s += v; q += v * v; }
        red[0][r][sub] = s; red[1][r][sub] = q;
    }
    __syncthreads();
    if ((tid & 15) == 0) {
        const int r = tid >> 4;
        float s = 0.f, q = 0.f;
        for (int i = 0; i < 16; ++i) { s += red[0][r][i]; q += red[1][r][i]; }
        const float m = s / (float)N;
        const float var = q / (float)N - m * m;
        mrow[r] = m;
        rrow[r] = 1.0f / sqrtf(var + 1e-5f);
    }
    __syncthreads();
    {
        const int r = tid >> 4, sub = tid & 15;
        const float m = mrow[r], rs = rrow[r];
        const size_t base = (size_t)(mTile * 16 + r) * N;
        for (int c = sub; c < N; c += 16) {
            float v = (st[r][c] - m) * rs;
            out[base + c] = f2bf(v > 0.f ? v : 0.f);  // ReLU + bf16
        }
    }
}

// ---------------------------------------------------------------------------
// 7) Layer-3 WMMA GEMM with fused per-16-row column partial sums (K=512,N=1024)
// ---------------------------------------------------------------------------
__global__ __launch_bounds__(256) void gemm_colsum_kernel(const unsigned short* __restrict__ A,
                                                          const unsigned short* __restrict__ Bt,
                                                          float* __restrict__ partial) {
    __shared__ float cs[8][2][16];
    const int tid = threadIdx.x;
    const int wave = tid >> 5, lane = tid & 31;
    const int h = lane >> 4, ln = lane & 15;
    const int mTile = blockIdx.x;
    const int ntile = blockIdx.y * 8 + wave;

    const unsigned short* arow = A + (size_t)(mTile * 16 + ln) * N2;
    const unsigned short* bbase = Bt + (size_t)(ntile * 16 + ln) * N2;

    v8f acc = {};
    for (int k0 = 0; k0 < N2; k0 += 32) {
        Frag32 af, bf;
        af.q[0] = *(const uint4*)(arow + k0 + 16 * h);
        af.q[1] = *(const uint4*)(arow + k0 + 16 * h + 8);
        bf.q[0] = *(const uint4*)(bbase + k0 + 16 * h);
        bf.q[1] = *(const uint4*)(bbase + k0 + 16 * h + 8);
        acc = __builtin_amdgcn_wmma_f32_16x16x32_bf16(
            false, af.v, false, bf.v, (short)0, acc, false, false);
    }
    float s = 0.f;
    #pragma unroll
    for (int r = 0; r < 8; ++r) s += acc[r];          // 8 rows of column ln
    cs[wave][h][ln] = s;
    __syncthreads();
    if (tid < 128) {                                  // combine the two halves
        const int w2 = tid >> 4, l2 = tid & 15;
        const float v = cs[w2][0][l2] + cs[w2][1][l2];
        partial[(size_t)mTile * N3 + (blockIdx.y * 8 + w2) * 16 + l2] = v;
    }
}

__global__ void colreduce_kernel(const float* __restrict__ partial,
                                 const float* __restrict__ b3,
                                 float* __restrict__ pooled) {
    const int c = blockIdx.x * 256 + threadIdx.x;
    if (c >= N3) return;
    float s = 0.f;
    for (int i = 0; i < MROWS / 16; ++i) s += partial[(size_t)i * N3 + c];
    pooled[c] = s / (float)MROWS + b3[c];             // bias folded after pool
}

// ---------------------------------------------------------------------------
// 8) Classifier head: 1x1024 -> 512 (GN,ReLU) -> 256 (GN,ReLU) -> 2
// ---------------------------------------------------------------------------
__global__ __launch_bounds__(512) void head_kernel(const float* __restrict__ pooled,
                                                   const float* __restrict__ Wc1, const float* __restrict__ bc1,
                                                   const float* __restrict__ g1,  const float* __restrict__ be1,
                                                   const float* __restrict__ Wc2, const float* __restrict__ bc2,
                                                   const float* __restrict__ g2,  const float* __restrict__ be2,
                                                   const float* __restrict__ Wc3, const float* __restrict__ bc3,
                                                   float* __restrict__ out) {
    __shared__ float p[1024], h1[512], h2[256], gm[32], gr[32];
    const int t = threadIdx.x;
    for (int i = t; i < 1024; i += 512) p[i] = pooled[i];
    __syncthreads();
    {   // c1 = pooled @ Wc1 + bc1
        float s = bc1[t];
        for (int k = 0; k < 1024; ++k) s += p[k] * Wc1[(size_t)k * 512 + t];
        h1[t] = s;
    }
    __syncthreads();
    if (t < 32) {   // group norm stats, G=32, group size 16
        float s = 0.f, q = 0.f;
        for (int i = 0; i < 16; ++i) { float v = h1[t * 16 + i]; s += v; q += v * v; }
        float m = s / 16.f;
        gm[t] = m; gr[t] = 1.0f / sqrtf(q / 16.f - m * m + 1e-5f);
    }
    __syncthreads();
    {
        const int g = t >> 4;
        float v = (h1[t] - gm[g]) * gr[g] * g1[t] + be1[t];
        h1[t] = v > 0.f ? v : 0.f;
    }
    __syncthreads();
    if (t < 256) {  // c2 = h1 @ Wc2 + bc2
        float s = bc2[t];
        for (int k = 0; k < 512; ++k) s += h1[k] * Wc2[(size_t)k * 256 + t];
        h2[t] = s;
    }
    __syncthreads();
    if (t < 32) {   // group norm, G=32, group size 8
        float s = 0.f, q = 0.f;
        for (int i = 0; i < 8; ++i) { float v = h2[t * 8 + i]; s += v; q += v * v; }
        float m = s / 8.f;
        gm[t] = m; gr[t] = 1.0f / sqrtf(q / 8.f - m * m + 1e-5f);
    }
    __syncthreads();
    if (t < 256) {
        const int g = t >> 3;
        float v = (h2[t] - gm[g]) * gr[g] * g2[t] + be2[t];
        h2[t] = v > 0.f ? v : 0.f;
    }
    __syncthreads();
    if (t < 2) {
        float s = bc3[t];
        for (int k = 0; k < 256; ++k) s += h2[k] * Wc3[(size_t)k * 2 + t];
        out[t] = s;
    }
}

// ---------------------------------------------------------------------------
// Launch
// ---------------------------------------------------------------------------
extern "C" void kernel_launch(void* const* d_in, const int* in_sizes, int n_in,
                              void* d_out, int out_size, void* d_ws, size_t ws_size,
                              hipStream_t stream) {
    const float* ref_pts = (const float*)d_in[0];
    const float* src_pts = (const float*)d_in[1];
    const float* ref_fts = (const float*)d_in[2];
    const float* src_fts = (const float*)d_in[3];
    const float* W1  = (const float*)d_in[4];   const float* b1  = (const float*)d_in[5];
    const float* W2  = (const float*)d_in[6];   const float* b2  = (const float*)d_in[7];
    const float* W3  = (const float*)d_in[8];   const float* b3  = (const float*)d_in[9];
    const float* Wc1 = (const float*)d_in[10];  const float* bc1 = (const float*)d_in[11];
    const float* g1  = (const float*)d_in[12];  const float* be1 = (const float*)d_in[13];
    const float* Wc2 = (const float*)d_in[14];  const float* bc2 = (const float*)d_in[15];
    const float* g2  = (const float*)d_in[16];  const float* be2 = (const float*)d_in[17];
    const float* Wc3 = (const float*)d_in[18];  const float* bc3 = (const float*)d_in[19];
    float* outp = (float*)d_out;

    // Workspace carve-up (~66.5 MB; L2-resident on MI455X's 192 MB L2)
    char* ws = (char*)d_ws;
    size_t off = 0;
    auto alloc = [&](size_t bytes) { char* p = ws + off; off += (bytes + 255) & ~(size_t)255; return p; };
    float*          rfN    = (float*)alloc((size_t)N_REF * DFEAT * 4);
    float*          sfN    = (float*)alloc((size_t)N_SRC * DFEAT * 4);
    float*          ptsN   = (float*)alloc((size_t)NPTS * 3 * 4);
    float*          dRef   = (float*)alloc(N_REF * 4);
    int*            iRef   = (int*)  alloc(N_REF * 4);
    float*          dSrc   = (float*)alloc(N_SRC * 4);
    int*            iSrc   = (int*)  alloc(N_SRC * 4);
    float*          partD  = (float*)alloc((size_t)NCHUNK * N_REF * 4);
    int*            partI  = (int*)  alloc((size_t)NCHUNK * N_REF * 4);
    unsigned short* featB  = (unsigned short*)alloc((size_t)MROWS * K1PAD * 2);
    unsigned short* Wt1    = (unsigned short*)alloc((size_t)N1 * K1PAD * 2);
    unsigned short* Wt2    = (unsigned short*)alloc((size_t)N2 * DFEAT * 2);
    unsigned short* Wt3    = (unsigned short*)alloc((size_t)N3 * N2 * 2);
    unsigned short* X1     = (unsigned short*)alloc((size_t)MROWS * N1 * 2);
    unsigned short* X2     = (unsigned short*)alloc((size_t)MROWS * N2 * 2);
    float*          partCS = (float*)alloc((size_t)(MROWS / 16) * N3 * 4);
    float*          pooled = (float*)alloc(N3 * 4);
    (void)ws_size; (void)in_sizes; (void)n_in; (void)out_size;

    // 1) feature L2 norm
    l2norm_kernel<<<N_REF, 256, 0, stream>>>(ref_fts, rfN);
    l2norm_kernel<<<N_SRC, 256, 0, stream>>>(src_fts, sfN);

    // 2) point normalization
    ptsnorm_kernel<<<1, 1024, 0, stream>>>(ref_pts, src_pts, ptsN);

    // 3) nearest neighbors both directions (chunked, then merged; sequential
    //    kernels on one stream so partD/partI are safely reused)
    minargmin_kernel<<<dim3(N_REF / 128, NCHUNK), 128, 0, stream>>>(ref_pts, src_pts, N_REF, N_SRC, partD, partI);
    minmerge_kernel<<<N_REF / 256, 256, 0, stream>>>(partD, partI, N_REF, dRef, iRef);
    minargmin_kernel<<<dim3(N_SRC / 128, NCHUNK), 128, 0, stream>>>(src_pts, ref_pts, N_SRC, N_REF, partD, partI);
    minmerge_kernel<<<N_SRC / 256, 256, 0, stream>>>(partD, partI, N_SRC, dSrc, iSrc);

    // 4) gather + scores + bf16 feat matrix (K padded to 544)
    featbuild_kernel<<<MROWS, 256, 0, stream>>>(rfN, sfN, ptsN, dRef, iRef, dSrc, iSrc, featB);

    // 5) weights -> bf16, transposed to N x K for contiguous B fragments
    wconv_kernel<<<((size_t)N1 * K1PAD + 255) / 256, 256, 0, stream>>>(W1, Wt1, K1RAW, K1PAD, N1);
    wconv_kernel<<<((size_t)N2 * DFEAT + 255) / 256, 256, 0, stream>>>(W2, Wt2, DFEAT, DFEAT, N2);
    wconv_kernel<<<((size_t)N3 * N2 + 255) / 256, 256, 0, stream>>>(W3, Wt3, N2, N2, N3);

    // 6) MLP on WMMA
    gemm_instnorm_relu<N1, K1PAD><<<MROWS / 16, 256, 0, stream>>>(featB, Wt1, b1, X1);
    gemm_instnorm_relu<N2, N1>  <<<MROWS / 16, 256, 0, stream>>>(X1, Wt2, b2, X2);
    gemm_colsum_kernel<<<dim3(MROWS / 16, N3 / 128), 256, 0, stream>>>(X2, Wt3, partCS);
    colreduce_kernel<<<N3 / 256, 256, 0, stream>>>(partCS, b3, pooled);

    // 7) head
    head_kernel<<<1, 512, 0, stream>>>(pooled, Wc1, bc1, g1, be1, Wc2, bc2, g2, be2, Wc3, bc3, outp);
}
